// Model_57260503990652
// MI455X (gfx1250) — matrix operand order repaired and checked by reading, not compiled, not run on hardware
//
#include <hip/hip_runtime.h>
#include <hip/hip_bf16.h>
#include <math.h>

// ---------------- problem constants (match reference) ----------------
#define N_NODES 131072      // B*L
#define N_EDGES 2097152
#define HDIM    96
#define DIN     300
#define NGRAPH  256
#define LNODES  512
#define NCLASS  2

typedef __bf16 bf16_t;
typedef __attribute__((ext_vector_type(16))) __bf16 v16bf;
typedef __attribute__((ext_vector_type(8)))  float  v8f;

union Frag16 {
    v16bf  v;
    uint4  q[2];
    __bf16 h[16];
};

// native f32 -> bf16 conversion (backend selects hardware cvt on gfx1250)
__device__ __forceinline__ __bf16 f2bf(float f) { return (__bf16)f; }

__device__ __forceinline__ float sigm(float x) { return 1.f / (1.f + __expf(-x)); }

__device__ __forceinline__ v8f wmma_bf16(const Frag16& a, const Frag16& b, v8f c) {
    return __builtin_amdgcn_wmma_f32_16x16x32_bf16(
        /*neg_a=*/false, a.v, /*neg_b=*/false, b.v,
        /*c_mod=*/(short)0, c, /*reuse_a=*/false, /*reuse_b=*/false);
}

// A-fragment (16x32 bf16) from row-major bf16 buffer, row stride HDIM.
// Lane layout (ISA 7.12.2): lanes 0-15 -> K = kb+hi*8 + [0..7] and kb+16+hi*8 + [0..7]
__device__ __forceinline__ Frag16 load_a_frag(const bf16_t* row_base, int kb, int hi) {
    Frag16 f;
    const uint4* p = (const uint4*)(row_base + kb + hi * 8); // 16B aligned
    f.q[0] = p[0];   // elements kb+hi*8    .. +7
    f.q[1] = p[2];   // elements kb+hi*8+16 .. +23
    return f;
}

// B-fragment from pre-packed weight: frag index -> 32 lanes x 16 bf16 contiguous.
__device__ __forceinline__ Frag16 load_b_frag(const bf16_t* wpack, int frag, int lane) {
    Frag16 f;
    const uint4* p = (const uint4*)wpack + ((size_t)frag * 32 + lane) * 2;
    f.q[0] = p[0];
    f.q[1] = p[1];
    return f;
}

// ---------------- weight packing: W[K x 96] f32 -> B-fragment bf16 layout ----
__global__ void k_pack_w(const float* __restrict__ w, bf16_t* __restrict__ dst, int K) {
    int frag = blockIdx.x;          // kc*6 + nt
    int kc = frag / 6, nt = frag % 6;
    int lane = threadIdx.x;         // 32 threads
    int n = nt * 16 + (lane & 15);
    int kbase = kc * 32 + (lane >> 4) * 8;
#pragma unroll
    for (int i = 0; i < 16; ++i) {
        int k = kbase + (i >> 3) * 16 + (i & 7);
        float v = (k < K) ? w[(size_t)k * HDIM + n] : 0.f;
        dst[((size_t)frag * 32 + lane) * 16 + i] = f2bf(v);
    }
}

// ---------------- encoder: x = tanh(embed[node_ids] @ enc_w + enc_b) --------
// one wave per 16-node tile; K padded 300 -> 320 (10 chunks, last one guarded)
__global__ void k_encoder(const int* __restrict__ node_ids,
                          const float* __restrict__ embed,
                          const bf16_t* __restrict__ wenc,
                          const float* __restrict__ enc_b,
                          float* __restrict__ x, bf16_t* __restrict__ x16) {
    int tile = blockIdx.x * (blockDim.x >> 5) + (threadIdx.x >> 5);
    int lane = threadIdx.x & 31;
    int m = lane & 15, hi = lane >> 4;
    int row = tile * 16 + m;
    const float* erow = embed + (size_t)node_ids[row] * DIN;

    v8f acc[6] = {};
    // fast k-chunks 0..8: all K indices < 300, vector loads
    for (int kc = 0; kc < 9; ++kc) {
        int kb = kc * 32;
        Frag16 fa;
        float buf[16];
        const float4* p0 = (const float4*)(erow + kb + hi * 8);        // K: kb+hi*8 .. +7
        const float4* p1 = (const float4*)(erow + kb + 16 + hi * 8);   // K: kb+16+hi*8 .. +7
        *(float4*)&buf[0]  = p0[0];
        *(float4*)&buf[4]  = p0[1];
        *(float4*)&buf[8]  = p1[0];
        *(float4*)&buf[12] = p1[1];
#pragma unroll
        for (int i = 0; i < 16; ++i) fa.h[i] = f2bf(buf[i]);
#pragma unroll
        for (int nt = 0; nt < 6; ++nt) {
            Frag16 fb = load_b_frag(wenc, kc * 6 + nt, lane);
            acc[nt] = wmma_bf16(fa, fb, acc[nt]);
        }
    }
    // final chunk kb=288: guarded (K 288..319, valid < 300)
    {
        const int kb = 288;
        Frag16 fa;
#pragma unroll
        for (int i = 0; i < 16; ++i) {
            int k = kb + (i >> 3) * 16 + hi * 8 + (i & 7);
            fa.h[i] = f2bf((k < DIN) ? erow[k] : 0.f);
        }
#pragma unroll
        for (int nt = 0; nt < 6; ++nt) {
            Frag16 fb = load_b_frag(wenc, 9 * 6 + nt, lane);
            acc[nt] = wmma_bf16(fa, fb, acc[nt]);
        }
    }
#pragma unroll
    for (int nt = 0; nt < 6; ++nt) {
        int ocol = nt * 16 + (lane & 15);
        float b = enc_b[ocol];
#pragma unroll
        for (int r = 0; r < 8; ++r) {
            int orow = tile * 16 + r + hi * 8;
            float v = tanhf(acc[nt][r] + b);
            x[(size_t)orow * HDIM + ocol] = v;
            x16[(size_t)orow * HDIM + ocol] = f2bf(v);
        }
    }
}

// ---------------- utility kernels ------------------------------------------
__global__ void k_zero4(float* __restrict__ p, int n4) {
    int i = blockIdx.x * blockDim.x + threadIdx.x;
    if (i < n4) ((float4*)p)[i] = make_float4(0.f, 0.f, 0.f, 0.f);
}
__global__ void k_cvt_bf16x4(const float* __restrict__ s, bf16_t* __restrict__ d, int n4) {
    int i = blockIdx.x * blockDim.x + threadIdx.x;
    if (i < n4) {
        float4 v = ((const float4*)s)[i];
        union { bf16_t h[4]; unsigned long long u; } r;
        r.h[0] = f2bf(v.x); r.h[1] = f2bf(v.y); r.h[2] = f2bf(v.z); r.h[3] = f2bf(v.w);
        ((unsigned long long*)d)[i] = r.u;
    }
}

// ---------------- edge scatter: a[dst] += x[src] * w -----------------------
__global__ void k_scatter(const int* __restrict__ ei, const float* __restrict__ ea,
                          const float* __restrict__ x, float* __restrict__ a) {
    int e = blockIdx.x * (blockDim.x >> 5) + (threadIdx.x >> 5);
    int lane = threadIdx.x & 31;
    if (e >= N_EDGES) return;
    int s = ei[e];
    int d = ei[N_EDGES + e];
    float w = ea[e];
    const float* xs = x + (size_t)s * HDIM;
    float* ad = a + (size_t)d * HDIM;
    atomicAdd(ad + lane,      xs[lane]      * w);
    atomicAdd(ad + lane + 32, xs[lane + 32] * w);
    atomicAdd(ad + lane + 64, xs[lane + 64] * w);
}

// ---------------- GRU gates: z = sig(a@z0 + x@z1 + b), r = sig(...) --------
// writes z (f32) and xr16 = bf16(x * r)
__global__ void k_gates(const bf16_t* __restrict__ a16, const bf16_t* __restrict__ x16,
                        const float* __restrict__ xf,
                        const bf16_t* __restrict__ wz0, const bf16_t* __restrict__ wz1,
                        const bf16_t* __restrict__ wr0, const bf16_t* __restrict__ wr1,
                        const float* __restrict__ z0b, const float* __restrict__ z1b,
                        const float* __restrict__ r0b, const float* __restrict__ r1b,
                        float* __restrict__ z, bf16_t* __restrict__ xr16) {
    int tile = blockIdx.x * (blockDim.x >> 5) + (threadIdx.x >> 5);
    int lane = threadIdx.x & 31;
    int m = lane & 15, hi = lane >> 4;
    const bf16_t* arow = a16 + (size_t)(tile * 16 + m) * HDIM;
    const bf16_t* xrow = x16 + (size_t)(tile * 16 + m) * HDIM;

    v8f accz[6] = {}, accr[6] = {};
    for (int kc = 0; kc < 3; ++kc) {
        Frag16 fa = load_a_frag(arow, kc * 32, hi);
        Frag16 fx = load_a_frag(xrow, kc * 32, hi);
#pragma unroll
        for (int nt = 0; nt < 6; ++nt) {
            int frag = kc * 6 + nt;
            Frag16 b0 = load_b_frag(wz0, frag, lane);
            accz[nt] = wmma_bf16(fa, b0, accz[nt]);
            Frag16 b1 = load_b_frag(wz1, frag, lane);
            accz[nt] = wmma_bf16(fx, b1, accz[nt]);
            Frag16 b2 = load_b_frag(wr0, frag, lane);
            accr[nt] = wmma_bf16(fa, b2, accr[nt]);
            Frag16 b3 = load_b_frag(wr1, frag, lane);
            accr[nt] = wmma_bf16(fx, b3, accr[nt]);
        }
    }
#pragma unroll
    for (int nt = 0; nt < 6; ++nt) {
        int ocol = nt * 16 + (lane & 15);
        float bz = z0b[ocol] + z1b[ocol];
        float br = r0b[ocol] + r1b[ocol];
#pragma unroll
        for (int r = 0; r < 8; ++r) {
            size_t idx = (size_t)(tile * 16 + r + hi * 8) * HDIM + ocol;
            float zv = sigm(accz[nt][r] + bz);
            float rv = sigm(accr[nt][r] + br);
            z[idx] = zv;
            xr16[idx] = f2bf(xf[idx] * rv);
        }
    }
}

// ---------------- GRU candidate + blend:
// h = tanh(a@h0 + (x*r)@h1 + b); x = h*z + x*(1-z)
__global__ void k_hstate(const bf16_t* __restrict__ a16, const bf16_t* __restrict__ xr16,
                         const bf16_t* __restrict__ wh0, const bf16_t* __restrict__ wh1,
                         const float* __restrict__ h0b, const float* __restrict__ h1b,
                         const float* __restrict__ z,
                         float* __restrict__ x, bf16_t* __restrict__ x16) {
    int tile = blockIdx.x * (blockDim.x >> 5) + (threadIdx.x >> 5);
    int lane = threadIdx.x & 31;
    int m = lane & 15, hi = lane >> 4;
    const bf16_t* arow  = a16  + (size_t)(tile * 16 + m) * HDIM;
    const bf16_t* xrrow = xr16 + (size_t)(tile * 16 + m) * HDIM;

    v8f acc[6] = {};
    for (int kc = 0; kc < 3; ++kc) {
        Frag16 fa  = load_a_frag(arow,  kc * 32, hi);
        Frag16 fxr = load_a_frag(xrrow, kc * 32, hi);
#pragma unroll
        for (int nt = 0; nt < 6; ++nt) {
            int frag = kc * 6 + nt;
            Frag16 b0 = load_b_frag(wh0, frag, lane);
            acc[nt] = wmma_bf16(fa, b0, acc[nt]);
            Frag16 b1 = load_b_frag(wh1, frag, lane);
            acc[nt] = wmma_bf16(fxr, b1, acc[nt]);
        }
    }
#pragma unroll
    for (int nt = 0; nt < 6; ++nt) {
        int ocol = nt * 16 + (lane & 15);
        float b = h0b[ocol] + h1b[ocol];
#pragma unroll
        for (int r = 0; r < 8; ++r) {
            size_t idx = (size_t)(tile * 16 + r + hi * 8) * HDIM + ocol;
            float hv = tanhf(acc[nt][r] + b);
            float zv = z[idx];
            float xv = x[idx];
            float nv = hv * zv + xv * (1.f - zv);
            x[idx] = nv;
            x16[idx] = f2bf(nv);
        }
    }
}

// ---------------- attention scalar per node: att = sig(x . att_w + b) ------
__global__ void k_att(const float* __restrict__ x, const float* __restrict__ att_w,
                      const float* __restrict__ att_b, float* __restrict__ att) {
    int n = blockIdx.x * blockDim.x + threadIdx.x;
    if (n >= N_NODES) return;
    const float4* xr = (const float4*)(x + (size_t)n * HDIM);
    const float4* wr = (const float4*)att_w;
    float s = 0.f;
#pragma unroll
    for (int i = 0; i < HDIM / 4; ++i) {
        float4 v = xr[i], w = wr[i];
        s += v.x * w.x + v.y * w.y + v.z * w.z + v.w * w.w;
    }
    att[n] = sigm(s + att_b[0]);
}

// ---------------- readout embedding: y = att * tanh(x @ emb_w + emb_b) -----
__global__ void k_emb(const bf16_t* __restrict__ x16, const bf16_t* __restrict__ wemb,
                      const float* __restrict__ emb_b, const float* __restrict__ att,
                      float* __restrict__ y) {
    int tile = blockIdx.x * (blockDim.x >> 5) + (threadIdx.x >> 5);
    int lane = threadIdx.x & 31;
    int m = lane & 15, hi = lane >> 4;
    const bf16_t* xrow = x16 + (size_t)(tile * 16 + m) * HDIM;

    v8f acc[6] = {};
    for (int kc = 0; kc < 3; ++kc) {
        Frag16 fx = load_a_frag(xrow, kc * 32, hi);
#pragma unroll
        for (int nt = 0; nt < 6; ++nt) {
            Frag16 fb = load_b_frag(wemb, kc * 6 + nt, lane);
            acc[nt] = wmma_bf16(fx, fb, acc[nt]);
        }
    }
#pragma unroll
    for (int nt = 0; nt < 6; ++nt) {
        int ocol = nt * 16 + (lane & 15);
        float b = emb_b[ocol];
#pragma unroll
        for (int r = 0; r < 8; ++r) {
            int orow = tile * 16 + r + hi * 8;
            y[(size_t)orow * HDIM + ocol] = att[orow] * tanhf(acc[nt][r] + b);
        }
    }
}

// ---------------- per-graph pooling + classifier ---------------------------
__global__ void k_pool(const float* __restrict__ y, const float* __restrict__ mlp_w,
                       const float* __restrict__ mlp_b, float* __restrict__ out) {
    __shared__ float pooled[HDIM];
    int g = blockIdx.x;
    int t = threadIdx.x;            // blockDim = 96
    if (t < HDIM) {
        const float* base = y + ((size_t)g * LNODES) * HDIM + t;
        float mx = -INFINITY, sm = 0.f;
        for (int l = 0; l < LNODES; ++l) {
            float v = base[(size_t)l * HDIM];
            mx = fmaxf(mx, v);
            sm += v;
        }
        pooled[t] = mx + sm * (1.f / (float)LNODES);
    }
    __syncthreads();
    if (t < NCLASS) {
        float s = mlp_b[t];
        for (int hcol = 0; hcol < HDIM; ++hcol)
            s += pooled[hcol] * mlp_w[hcol * NCLASS + t];
        out[g * NCLASS + t] = s;
    }
}

// ---------------- host-side launcher ---------------------------------------
extern "C" void kernel_launch(void* const* d_in, const int* in_sizes, int n_in,
                              void* d_out, int out_size, void* d_ws, size_t ws_size,
                              hipStream_t stream) {
    const int*   node_ids = (const int*)d_in[0];
    const int*   edge_idx = (const int*)d_in[1];
    const float* edge_att = (const float*)d_in[2];
    const float* embed    = (const float*)d_in[3];
    const float* enc_w = (const float*)d_in[4],  *enc_b = (const float*)d_in[5];
    const float* z0_w  = (const float*)d_in[6],  *z0_b  = (const float*)d_in[7];
    const float* z1_w  = (const float*)d_in[8],  *z1_b  = (const float*)d_in[9];
    const float* r0_w  = (const float*)d_in[10], *r0_b  = (const float*)d_in[11];
    const float* r1_w  = (const float*)d_in[12], *r1_b  = (const float*)d_in[13];
    const float* h0_w  = (const float*)d_in[14], *h0_b  = (const float*)d_in[15];
    const float* h1_w  = (const float*)d_in[16], *h1_b  = (const float*)d_in[17];
    const float* att_w = (const float*)d_in[18], *att_b = (const float*)d_in[19];
    const float* emb_w = (const float*)d_in[20], *emb_b = (const float*)d_in[21];
    const float* mlp_w = (const float*)d_in[22], *mlp_b = (const float*)d_in[23];
    float* out = (float*)d_out;

    // ---- workspace layout ----
    const size_t NH  = (size_t)N_NODES * HDIM;            // 12,582,912
    char* ws = (char*)d_ws;
    float*  x    = (float*)(ws);                          // NH f32
    float*  a    = (float*)(ws + NH * 4);                 // NH f32 (also y)
    float*  z    = (float*)(ws + NH * 8);                 // NH f32
    bf16_t* x16  = (bf16_t*)(ws + NH * 12);               // NH bf16
    bf16_t* a16  = (bf16_t*)(ws + NH * 14);               // NH bf16
    bf16_t* xr16 = (bf16_t*)(ws + NH * 16);               // NH bf16
    float*  att  = (float*)(ws + NH * 18);                // N f32
    bf16_t* wpk  = (bf16_t*)(ws + NH * 18 + (size_t)N_NODES * 4);
    const size_t FRAG_ELEMS = 512;                        // 32 lanes * 16 bf16
    bf16_t* wp_enc = wpk;                                 // 60 frags
    bf16_t* wp_z0  = wp_enc + 60 * FRAG_ELEMS;            // 18 frags each below
    bf16_t* wp_z1  = wp_z0  + 18 * FRAG_ELEMS;
    bf16_t* wp_r0  = wp_z1  + 18 * FRAG_ELEMS;
    bf16_t* wp_r1  = wp_r0  + 18 * FRAG_ELEMS;
    bf16_t* wp_h0  = wp_r1  + 18 * FRAG_ELEMS;
    bf16_t* wp_h1  = wp_h0  + 18 * FRAG_ELEMS;
    bf16_t* wp_emb = wp_h1  + 18 * FRAG_ELEMS;

    // ---- pack weights into WMMA B-fragment layout ----
    hipLaunchKernelGGL(k_pack_w, dim3(60), dim3(32), 0, stream, enc_w, wp_enc, DIN);
    hipLaunchKernelGGL(k_pack_w, dim3(18), dim3(32), 0, stream, z0_w, wp_z0, HDIM);
    hipLaunchKernelGGL(k_pack_w, dim3(18), dim3(32), 0, stream, z1_w, wp_z1, HDIM);
    hipLaunchKernelGGL(k_pack_w, dim3(18), dim3(32), 0, stream, r0_w, wp_r0, HDIM);
    hipLaunchKernelGGL(k_pack_w, dim3(18), dim3(32), 0, stream, r1_w, wp_r1, HDIM);
    hipLaunchKernelGGL(k_pack_w, dim3(18), dim3(32), 0, stream, h0_w, wp_h0, HDIM);
    hipLaunchKernelGGL(k_pack_w, dim3(18), dim3(32), 0, stream, h1_w, wp_h1, HDIM);
    hipLaunchKernelGGL(k_pack_w, dim3(18), dim3(32), 0, stream, emb_w, wp_emb, HDIM);

    const int TILES = N_NODES / 16;          // 8192
    const int WPB   = 8;                     // waves (tiles) per 256-thread block
    dim3 gemm_grid(TILES / WPB), gemm_blk(256);
    const int n4 = (int)(NH / 4);
    dim3 v4_grid((n4 + 255) / 256), v4_blk(256);

    // ---- encoder ----
    hipLaunchKernelGGL(k_encoder, gemm_grid, gemm_blk, 0, stream,
                       node_ids, embed, wp_enc, enc_b, x, x16);

    // ---- message passing steps ----
    for (int step = 0; step < 2; ++step) {
        hipLaunchKernelGGL(k_zero4, v4_grid, v4_blk, 0, stream, a, n4);
        hipLaunchKernelGGL(k_scatter, dim3(N_EDGES / 8), dim3(256), 0, stream,
                           edge_idx, edge_att, x, a);
        hipLaunchKernelGGL(k_cvt_bf16x4, v4_grid, v4_blk, 0, stream, a, a16, n4);
        hipLaunchKernelGGL(k_gates, gemm_grid, gemm_blk, 0, stream,
                           a16, x16, x, wp_z0, wp_z1, wp_r0, wp_r1,
                           z0_b, z1_b, r0_b, r1_b, z, xr16);
        hipLaunchKernelGGL(k_hstate, gemm_grid, gemm_blk, 0, stream,
                           a16, xr16, wp_h0, wp_h1, h0_b, h1_b, z, x, x16);
    }

    // ---- readout ----
    hipLaunchKernelGGL(k_att, dim3((N_NODES + 255) / 256), dim3(256), 0, stream,
                       x, att_w, att_b, att);
    hipLaunchKernelGGL(k_emb, gemm_grid, gemm_blk, 0, stream,
                       x16, wp_emb, emb_b, att, /*y=*/a);
    hipLaunchKernelGGL(k_pool, dim3(NGRAPH), dim3(96), 0, stream,
                       /*y=*/a, mlp_w, mlp_b, out);
}
